// GPT_57432302682355
// MI455X (gfx1250) — compile-verified, matrix operand order checked by reading, not run on hardware
//
#include <hip/hip_runtime.h>
#include <math.h>

// GPT-2 forward (B=2,T=512,C=768,H=12,L=12,V=50304) + distance head.
// GEMMs: f16 operands, v_wmma_f32_16x16x32_f16, wave32.
// Big GEMMs: 128-thread block = 64x64 tile, shared B panel double-buffered in
// LDS via global_load_async_to_lds_b128 (ASYNCcnt), 4 waves x 16x64 strips.

typedef __attribute__((ext_vector_type(16))) _Float16 v16h;
typedef __attribute__((ext_vector_type(8)))  float    v8f;
typedef _Float16 h16_t;

#define DEV __device__ __forceinline__

namespace {

constexpr int Tt  = 512;
constexpr int Cc  = 768;
constexpr int NH  = 12;
constexpr int HDd = 64;
constexpr int Vv  = 50304;
constexpr int Ln  = 12;
constexpr int Bn  = 2;
constexpr int BT  = Bn * Tt;        // 1024
constexpr int C3  = 3 * Cc;         // 2304
constexpr int C4  = 4 * Cc;         // 3072

constexpr int PANEL_H = 2048;       // halves per LDS panel buffer (64 rows x 32 k)

DEV float fnum(float v, float rep) {
  return (__builtin_isnan(v) || __builtin_isinf(v)) ? rep : v;
}

DEV float wred_sum(float v) {
  #pragma unroll
  for (int o = 16; o > 0; o >>= 1) v += __shfl_xor(v, o, 32);
  return v;
}
DEV float wred_max(float v) {
  #pragma unroll
  for (int o = 16; o > 0; o >>= 1) v = fmaxf(v, __shfl_xor(v, o, 32));
  return v;
}

// One 16-element f16 fragment slice for this lane (ISA 16-bit operand layout):
// caller adds hi*8; elements 0..7 <- p[0..7], elements 8..15 <- p[16..23].
DEV v16h load_h16(const h16_t* __restrict__ p) {
  union { v16h v; float4 f[2]; } u;
  u.f[0] = *(const float4*)(p);
  u.f[1] = *(const float4*)(p + 16);
  return u.v;
}

DEV v8f wmma16(v16h a, v16h b, v8f c) {
  return __builtin_amdgcn_wmma_f32_16x16x32_f16(false, a, false, b, (short)0, c,
                                                false, false);
}

// Async DMA: two 16-byte lane-chunks global -> LDS (tracked by ASYNCcnt).
DEV void async_copy2(unsigned l0, unsigned l1, const h16_t* g0, const h16_t* g1) {
  asm volatile(
      "global_load_async_to_lds_b128 %0, %2, off\n\t"
      "global_load_async_to_lds_b128 %1, %3, off"
      :: "v"(l0), "v"(l1), "v"(g0), "v"(g1)
      : "memory");
}
DEV void wait_async0() { asm volatile("s_wait_asynccnt 0x0" ::: "memory"); }

// Direct-load strip core (attention): 16(M) x 64(N), A fragment reused x4.
DEV void mma_strip(const h16_t* __restrict__ ap, const h16_t* __restrict__ bp,
                   int ldw, int K, v8f (&acc)[4]) {
  for (int k0 = 0; k0 < K; k0 += 32) {
    v16h a = load_h16(ap + k0);
    #pragma unroll
    for (int j = 0; j < 4; ++j) {
      v16h b = load_h16(bp + (size_t)j * 16 * ldw + k0);
      acc[j] = wmma16(a, b, acc[j]);
    }
  }
}

// LDS-staged strip core: block-shared 64-row B panel, statically double-
// buffered (K multiple of 64), fed by async DMA. Each wave computes its own
// 16x64 strip against the shared panel. All 128 threads participate.
DEV void mma_strip_staged(const h16_t* __restrict__ ap,
                          const h16_t* __restrict__ wrow, int ldw, int K,
                          h16_t* smem, int tid, int row, int hi,
                          v8f (&acc)[4]) {
  // This thread's two 16B chunks of the 4KB panel: chunk c -> row c>>2, k (c&3)*8
  const int c0 = tid, c1 = tid + 128;
  const h16_t* g0 = wrow + (size_t)(c0 >> 2) * ldw + (c0 & 3) * 8;
  const h16_t* g1 = wrow + (size_t)(c1 >> 2) * ldw + (c1 & 3) * 8;
  const unsigned lbase = (unsigned)(uintptr_t)smem;  // LDS byte offset
  const unsigned l0 = lbase + (unsigned)((c0 >> 2) * 64 + (c0 & 3) * 16);
  const unsigned l1 = lbase + (unsigned)((c1 >> 2) * 64 + (c1 & 3) * 16);
  constexpr unsigned BUF = (unsigned)(PANEL_H * 2);  // 4096 bytes

  async_copy2(l0, l1, g0, g1);                    // prime buf0 with panel k=0
  for (int k0 = 0; k0 < K; k0 += 64) {
    // --- first half: compute buf0 (panel k0), prefetch panel k0+32 -> buf1 ---
    wait_async0();
    __syncthreads();
    async_copy2(l0 + BUF, l1 + BUF, g0 + k0 + 32, g1 + k0 + 32);
    {
      v16h a = load_h16(ap + k0);
      #pragma unroll
      for (int j = 0; j < 4; ++j) {
        const h16_t* lb = smem + (j * 16 + row) * 32 + hi * 8;
        acc[j] = wmma16(a, load_h16(lb), acc[j]);
      }
    }
    // --- second half: compute buf1 (panel k0+32), prefetch k0+64 -> buf0 ---
    wait_async0();
    __syncthreads();
    if (k0 + 64 < K)
      async_copy2(l0, l1, g0 + k0 + 64, g1 + k0 + 64);
    {
      v16h a = load_h16(ap + k0 + 32);
      #pragma unroll
      for (int j = 0; j < 4; ++j) {
        const h16_t* lb = smem + PANEL_H + (j * 16 + row) * 32 + hi * 8;
        acc[j] = wmma16(a, load_h16(lb), acc[j]);
      }
    }
  }
}

// ---------------------------------------------------------------------------
// fp32 -> f16 pack (n multiple of 4)
__global__ void cvt_f16(const float* __restrict__ s, h16_t* __restrict__ d, int n) {
  int i = (blockIdx.x * 256 + threadIdx.x) * 4;
  if (i >= n) return;
  float4 v = *(const float4*)(s + i);
  union { h16_t h[4]; float2 f; } u;
  u.h[0] = (h16_t)v.x; u.h[1] = (h16_t)v.y; u.h[2] = (h16_t)v.z; u.h[3] = (h16_t)v.w;
  *(float2*)(d + i) = u.f;
}

// ---------------------------------------------------------------------------
__global__ void embed_k(const int* __restrict__ idx, const float* __restrict__ wte,
                        const float* __restrict__ wpe, float* __restrict__ x) {
  int i = blockIdx.x * 256 + threadIdx.x;
  if (i >= BT * Cc) return;
  int r = i / Cc, c = i - r * Cc;
  int t = r % Tt;
  x[i] = wte[(size_t)idx[r] * Cc + c] + wpe[(size_t)t * Cc + c];
}

// ---------------------------------------------------------------------------
// LayerNorm (C=768), one wave per row, fp32 in -> f16 out.
__global__ void layernorm_rows(const float* __restrict__ x, const float* __restrict__ w,
                               const float* __restrict__ b, h16_t* __restrict__ out,
                               int nrows) {
  int lane = threadIdx.x & 31;
  int r = blockIdx.x * (blockDim.x >> 5) + (threadIdx.x >> 5);
  if (r >= nrows) return;
  const float* xr = x + (size_t)r * Cc;
  float v[24];
  float s = 0.f;
  #pragma unroll
  for (int i = 0; i < 24; ++i) { v[i] = xr[lane + 32 * i]; s += v[i]; }
  s = wred_sum(s);
  float mean = s * (1.0f / Cc);
  float s2 = 0.f;
  #pragma unroll
  for (int i = 0; i < 24; ++i) { float d = v[i] - mean; s2 += d * d; }
  s2 = wred_sum(s2);
  float rstd = rsqrtf(s2 * (1.0f / Cc) + 1e-5f);
  h16_t* orow = out + (size_t)r * Cc;
  #pragma unroll
  for (int i = 0; i < 24; ++i) {
    int c = lane + 32 * i;
    orow[c] = (h16_t)((v[i] - mean) * rstd * w[c] + b[c]);
  }
}

// ---------------------------------------------------------------------------
// Staged NT GEMM, f16 output: out = (f16) fn(A.W^T + bias)     [qkv, fc]
// Grid: (M/64)*(N/64) blocks of 128. All dims divide exactly (barrier-safe).
__global__ void gemm_nt_f16o(const h16_t* __restrict__ A, int lda,
                             const h16_t* __restrict__ W, int ldw,
                             const float* __restrict__ bias,
                             h16_t* __restrict__ out, int ldo, int M, int N, int K) {
  __shared__ h16_t smem[2 * PANEL_H];
  int tid = threadIdx.x;
  int lane = tid & 31, w = tid >> 5;
  int strips = N >> 6;
  int bm = blockIdx.x / strips, ts = blockIdx.x - bm * strips;
  int row = lane & 15, hi = lane >> 4;
  int tm = bm * 4 + w;
  const h16_t* ap = A + (size_t)(tm * 16 + row) * lda + hi * 8;
  const h16_t* wrow = W + (size_t)(ts * 64) * ldw;
  v8f acc[4] = {{}, {}, {}, {}};
  mma_strip_staged(ap, wrow, ldw, K, smem, tid, row, hi, acc);
  int m0 = tm * 16 + hi * 8;
  #pragma unroll
  for (int j = 0; j < 4; ++j) {
    int n = ts * 64 + j * 16 + row;
    float bv = bias[n];
    #pragma unroll
    for (int i = 0; i < 8; ++i)
      out[(size_t)(m0 + i) * ldo + n] = (h16_t)fnum(acc[j][i] + bv, 0.0f);
  }
}

// Staged NT GEMM, fp32 residual: out = fn(res + fn(A.W^T + bias))  [proj, fc2]
__global__ void gemm_nt_res(const h16_t* __restrict__ A, int lda,
                            const h16_t* __restrict__ W, int ldw,
                            const float* __restrict__ bias,
                            const float* __restrict__ res,
                            float* __restrict__ out, int ldo, int M, int N, int K) {
  __shared__ h16_t smem[2 * PANEL_H];
  int tid = threadIdx.x;
  int lane = tid & 31, w = tid >> 5;
  int strips = N >> 6;
  int bm = blockIdx.x / strips, ts = blockIdx.x - bm * strips;
  int row = lane & 15, hi = lane >> 4;
  int tm = bm * 4 + w;
  const h16_t* ap = A + (size_t)(tm * 16 + row) * lda + hi * 8;
  const h16_t* wrow = W + (size_t)(ts * 64) * ldw;
  v8f acc[4] = {{}, {}, {}, {}};
  mma_strip_staged(ap, wrow, ldw, K, smem, tid, row, hi, acc);
  int m0 = tm * 16 + hi * 8;
  #pragma unroll
  for (int j = 0; j < 4; ++j) {
    int n = ts * 64 + j * 16 + row;
    float bv = bias[n];
    #pragma unroll
    for (int i = 0; i < 8; ++i) {
      float vv = fnum(acc[j][i] + bv, 0.0f);
      vv = fnum(res[(size_t)(m0 + i) * ldo + n] + vv, 0.0f);
      out[(size_t)(m0 + i) * ldo + n] = vv;
    }
  }
}

// ---------------------------------------------------------------------------
// att[b,h,q,k] = (f16) fn(scale * q.k); strips fully above diagonal skipped.
__global__ void attn_qk(const h16_t* __restrict__ qkv16, h16_t* __restrict__ att16,
                        float scale) {
  int lane = threadIdx.x & 31;
  int tile = blockIdx.x * (blockDim.x >> 5) + (threadIdx.x >> 5);
  constexpr int TM = Tt / 16, TS = Tt / 64;  // 32 x 8 strips per (b,h)
  if (tile >= Bn * NH * TM * TS) return;
  int bh = tile / (TM * TS);
  int r = tile - bh * (TM * TS);
  int tm = r / TS, ts = r - tm * TS;
  if (4 * ts > tm) return;  // strip entirely masked; softmax rewrites all cols
  int b = bh / NH, h = bh - b * NH;
  int row = lane & 15, hi = lane >> 4;
  const h16_t* ap = qkv16 + (size_t)(b * Tt + tm * 16 + row) * C3 + h * HDd + hi * 8;
  const h16_t* bp = qkv16 + (size_t)(b * Tt + ts * 64 + row) * C3 + Cc + h * HDd + hi * 8;
  v8f acc[4] = {{}, {}, {}, {}};
  mma_strip(ap, bp, C3, HDd, acc);
  h16_t* obase = att16 + (size_t)bh * Tt * Tt;
  int m0 = tm * 16 + hi * 8;
  #pragma unroll
  for (int j = 0; j < 4; ++j) {
    int n = ts * 64 + j * 16 + row;
    #pragma unroll
    for (int i = 0; i < 8; ++i)
      obase[(size_t)(m0 + i) * Tt + n] = (h16_t)fnum(acc[j][i] * scale, 0.0f);
  }
}

// ---------------------------------------------------------------------------
// Causal softmax over f16 scores in place (math in fp32); masked cols -> 0.
__global__ void softmax_causal(h16_t* __restrict__ att16) {
  int lane = threadIdx.x & 31;
  int r = blockIdx.x * (blockDim.x >> 5) + (threadIdx.x >> 5);
  if (r >= Bn * NH * Tt) return;
  int t = r % Tt;
  h16_t* rowp = att16 + (size_t)r * Tt;
  float mx = -3.4e38f;
  #pragma unroll
  for (int i = 0; i < 16; ++i) {
    int c = lane + 32 * i;
    if (c <= t) mx = fmaxf(mx, (float)rowp[c]);
  }
  mx = wred_max(mx);
  float vals[16];
  float sum = 0.f;
  #pragma unroll
  for (int i = 0; i < 16; ++i) {
    int c = lane + 32 * i;
    float e = (c <= t) ? __expf((float)rowp[c] - mx) : 0.0f;
    vals[i] = e;
    sum += e;
  }
  sum = wred_sum(sum);
  float inv = 1.0f / sum;
  #pragma unroll
  for (int i = 0; i < 16; ++i) {
    int c = lane + 32 * i;
    rowp[c] = (h16_t)fnum(vals[i] * inv, 0.0f);
  }
}

// ---------------------------------------------------------------------------
// vT[bh, d, t] = qkv16[b, t, 2C + h*HD + d]
__global__ void transpose_v(const h16_t* __restrict__ qkv16, h16_t* __restrict__ vT) {
  int i = blockIdx.x * 256 + threadIdx.x;
  if (i >= Bn * NH * HDd * Tt) return;
  int t = i & (Tt - 1);
  int rest = i >> 9;
  int d = rest & (HDd - 1);
  int bh = rest >> 6;
  int b = bh / NH, h = bh - b * NH;
  vT[i] = qkv16[(size_t)(b * Tt + t) * C3 + 2 * Cc + h * HDd + d];
}

// y[b,q,h,:] = (f16) fn(att . vT^T); K loop clipped at the causal diagonal.
__global__ void attn_av(const h16_t* __restrict__ att16, const h16_t* __restrict__ vT,
                        h16_t* __restrict__ y16) {
  int lane = threadIdx.x & 31;
  int tile = blockIdx.x * (blockDim.x >> 5) + (threadIdx.x >> 5);
  constexpr int TM = Tt / 16;
  if (tile >= Bn * NH * TM) return;
  int bh = tile / TM, tm = tile - bh * TM;
  int b = bh / NH, h = bh - b * NH;
  int row = lane & 15, hi = lane >> 4;
  const h16_t* ap = att16 + (size_t)bh * Tt * Tt + (size_t)(tm * 16 + row) * Tt + hi * 8;
  const h16_t* bp = vT + (size_t)bh * HDd * Tt + (size_t)row * Tt + hi * 8;
  v8f acc[4] = {{}, {}, {}, {}};
  int klim = ((tm + 1) * 16 + 31) & ~31;  // probs exactly 0 past the diagonal
  mma_strip(ap, bp, Tt, klim, acc);
  int m0 = tm * 16 + hi * 8;
  #pragma unroll
  for (int j = 0; j < 4; ++j) {
    int n = j * 16 + row;
    #pragma unroll
    for (int i = 0; i < 8; ++i)
      y16[(size_t)(b * Tt + m0 + i) * Cc + h * HDd + n] = (h16_t)fnum(acc[j][i], 0.0f);
  }
}

// ---------------------------------------------------------------------------
__global__ void gelu_k(h16_t* __restrict__ h, int n) {
  int i = blockIdx.x * 256 + threadIdx.x;
  if (i >= n) return;
  float xv = (float)h[i];
  float g = 0.5f * xv * (1.0f + tanhf(0.7978845608f * (xv + 0.044715f * xv * xv * xv)));
  h[i] = (h16_t)fnum(g, 0.0f);
}

// ---------------------------------------------------------------------------
template <typename Tsrc>
__global__ void rowsumsq(const Tsrc* __restrict__ a, float* __restrict__ o, int nrows) {
  int lane = threadIdx.x & 31;
  int r = blockIdx.x * (blockDim.x >> 5) + (threadIdx.x >> 5);
  if (r >= nrows) return;
  const Tsrc* rp = a + (size_t)r * Cc;
  float s = 0.f;
  #pragma unroll
  for (int i = 0; i < 24; ++i) { float v = (float)rp[lane + 32 * i]; s += v * v; }
  s = wred_sum(s);
  if (lane == 0) o[r] = fnum(s, 1.0f);
}

// ---------------------------------------------------------------------------
// Staged head GEMM + dist epilogue -> d_out
__global__ void gemm_dist(const h16_t* __restrict__ A, const h16_t* __restrict__ Wt,
                          const float* __restrict__ ww, const float* __restrict__ xx,
                          float* __restrict__ out) {
  __shared__ h16_t smem[2 * PANEL_H];
  int tid = threadIdx.x;
  int lane = tid & 31, w = tid >> 5;
  constexpr int strips = Vv / 64;  // 786
  int bm = blockIdx.x / strips, ts = blockIdx.x - bm * strips;
  int row = lane & 15, hi = lane >> 4;
  int tm = bm * 4 + w;
  const h16_t* ap = A + (size_t)(tm * 16 + row) * Cc + hi * 8;
  const h16_t* wrow = Wt + (size_t)(ts * 64) * Cc;
  v8f acc[4] = {{}, {}, {}, {}};
  mma_strip_staged(ap, wrow, Cc, Cc, smem, tid, row, hi, acc);
  int m0 = tm * 16 + hi * 8;
  #pragma unroll
  for (int j = 0; j < 4; ++j) {
    int n = ts * 64 + j * 16 + row;
    float wwn = ww[n];
    #pragma unroll
    for (int i = 0; i < 8; ++i) {
      float wx = fnum(acc[j][i], 0.0f);
      float d = wwn + xx[m0 + i] - 2.0f * wx;
      d = fnum(d, 768.0f);
      d = fnum(d * (1.0f / Cc), 1.0f);
      out[(size_t)(m0 + i) * Vv + n] = d;
    }
  }
}

// ---------------------------------------------------------------------------
__global__ void row_min(const float* __restrict__ d, float* __restrict__ o, int ncols) {
  __shared__ float sm[256];
  const float* rowp = d + (size_t)blockIdx.x * ncols;
  float m = 3.4e38f;
  for (int c = threadIdx.x; c < ncols; c += 256) m = fminf(m, rowp[c]);
  sm[threadIdx.x] = m;
  __syncthreads();
  for (int s = 128; s > 0; s >>= 1) {
    if ((int)threadIdx.x < s) sm[threadIdx.x] = fminf(sm[threadIdx.x], sm[threadIdx.x + s]);
    __syncthreads();
  }
  if (threadIdx.x == 0) o[blockIdx.x] = sm[0];
}

__global__ void row_sum(const float* __restrict__ d, float* __restrict__ o, int ncols) {
  __shared__ float sm[256];
  const float* rowp = d + (size_t)blockIdx.x * ncols;
  float s = 0.f;
  for (int c = threadIdx.x; c < ncols; c += 256) s += rowp[c];
  sm[threadIdx.x] = s;
  __syncthreads();
  for (int st = 128; st > 0; st >>= 1) {
    if ((int)threadIdx.x < st) sm[threadIdx.x] += sm[threadIdx.x + st];
    __syncthreads();
  }
  if (threadIdx.x == 0) o[blockIdx.x] = sm[0];
}

__global__ void dist_pow(float* __restrict__ d, const float* __restrict__ rmin) {
  size_t total = (size_t)BT * Vv;
  for (size_t i = (size_t)blockIdx.x * 256 + threadIdx.x; i < total;
       i += (size_t)gridDim.x * 256) {
    int m = (int)(i / Vv);
    float r = d[i] / rmin[m];
    r = fnum(r, 1.0f);
    float p = powf(r, -768.0f);
    d[i] = fnum(p, 1.0f);
  }
}

__global__ void logits_k(float* __restrict__ d, const float* __restrict__ rsum) {
  size_t total = (size_t)BT * Vv;
  for (size_t i = (size_t)blockIdx.x * 256 + threadIdx.x; i < total;
       i += (size_t)gridDim.x * 256) {
    int m = (int)(i / Vv);
    float pr = fnum(d[i] / rsum[m], 0.0f) + (0.01f / 50304.0f);
    d[i] = fnum(logf(pr), 0.0f);
  }
}

inline int staged_blocks(int M, int N) { return (M / 64) * (N / 64); }
inline int cvt_blocks(int n) { return (n / 4 + 255) / 256; }

}  // namespace

extern "C" void kernel_launch(void* const* d_in, const int* in_sizes, int n_in,
                              void* d_out, int out_size, void* d_ws, size_t ws_size,
                              hipStream_t stream) {
  (void)in_sizes; (void)n_in; (void)out_size; (void)ws_size;

  const int*   idx    = (const int*)  d_in[0];
  const float* wte    = (const float*)d_in[1];
  const float* wpe    = (const float*)d_in[2];
  const float* ln1_w  = (const float*)d_in[3];
  const float* ln1_b  = (const float*)d_in[4];
  const float* attn_w = (const float*)d_in[5];
  const float* attn_b = (const float*)d_in[6];
  const float* proj_w = (const float*)d_in[7];
  const float* proj_b = (const float*)d_in[8];
  const float* ln2_w  = (const float*)d_in[9];
  const float* ln2_b  = (const float*)d_in[10];
  const float* fc_w   = (const float*)d_in[11];
  const float* fc_b   = (const float*)d_in[12];
  const float* fc2_w  = (const float*)d_in[13];
  const float* fc2_b  = (const float*)d_in[14];
  const float* lnf_w  = (const float*)d_in[15];
  const float* lnf_b  = (const float*)d_in[16];
  float* out = (float*)d_out;

  // Workspace partition (~112 MB)
  char* p = (char*)d_ws;
  float*  x     = (float*)p;  p += (size_t)BT * Cc * 4;            // fp32 trunk
  h16_t*  h16   = (h16_t*)p;  p += (size_t)BT * Cc * 2;            // LN out
  h16_t*  qkv16 = (h16_t*)p;  p += (size_t)BT * C3 * 2;
  h16_t*  att16 = (h16_t*)p;  p += (size_t)Bn * NH * Tt * Tt * 2;  // scores->probs
  h16_t*  vT16  = (h16_t*)p;  p += (size_t)Bn * NH * HDd * Tt * 2;
  h16_t*  y16   = (h16_t*)p;  p += (size_t)BT * Cc * 2;
  h16_t*  hid16 = (h16_t*)p;  p += (size_t)BT * C4 * 2;
  h16_t*  wbuf  = (h16_t*)p;  p += (size_t)C4 * Cc * 2;            // per-layer W panel
  h16_t*  wte16 = (h16_t*)p;  p += (size_t)Vv * Cc * 2;
  float*  ww    = (float*)p;  p += (size_t)Vv * 4;
  float*  xx    = (float*)p;  p += (size_t)BT * 4;
  float*  rmin  = (float*)p;  p += (size_t)BT * 4;
  float*  rsum  = (float*)p;

  embed_k<<<(BT * Cc + 255) / 256, 256, 0, stream>>>(idx, wte, wpe, x);
  cvt_f16<<<cvt_blocks(Vv * Cc), 256, 0, stream>>>(wte, wte16, Vv * Cc);

  for (int l = 0; l < Ln; ++l) {
    // --- attention block ---
    layernorm_rows<<<128, 256, 0, stream>>>(x, ln1_w + l * Cc, ln1_b + l * Cc, h16, BT);
    cvt_f16<<<cvt_blocks(C3 * Cc), 256, 0, stream>>>(attn_w + (size_t)l * C3 * Cc, wbuf, C3 * Cc);
    gemm_nt_f16o<<<staged_blocks(BT, C3), 128, 0, stream>>>(
        h16, Cc, wbuf, Cc, attn_b + (size_t)l * C3, qkv16, C3, BT, C3, Cc);
    attn_qk<<<(Bn * NH * 32 * 8 + 3) / 4, 128, 0, stream>>>(qkv16, att16, 0.125f);
    softmax_causal<<<(Bn * NH * Tt + 7) / 8, 256, 0, stream>>>(att16);
    transpose_v<<<(Bn * NH * HDd * Tt + 255) / 256, 256, 0, stream>>>(qkv16, vT16);
    attn_av<<<(Bn * NH * 32 + 3) / 4, 128, 0, stream>>>(att16, vT16, y16);
    cvt_f16<<<cvt_blocks(Cc * Cc), 256, 0, stream>>>(proj_w + (size_t)l * Cc * Cc, wbuf, Cc * Cc);
    gemm_nt_res<<<staged_blocks(BT, Cc), 128, 0, stream>>>(
        y16, Cc, wbuf, Cc, proj_b + (size_t)l * Cc, x, x, Cc, BT, Cc, Cc);
    // --- MLP block ---
    layernorm_rows<<<128, 256, 0, stream>>>(x, ln2_w + l * Cc, ln2_b + l * Cc, h16, BT);
    cvt_f16<<<cvt_blocks(C4 * Cc), 256, 0, stream>>>(fc_w + (size_t)l * C4 * Cc, wbuf, C4 * Cc);
    gemm_nt_f16o<<<staged_blocks(BT, C4), 128, 0, stream>>>(
        h16, Cc, wbuf, Cc, fc_b + (size_t)l * C4, hid16, C4, BT, C4, Cc);
    gelu_k<<<(BT * C4 + 255) / 256, 256, 0, stream>>>(hid16, BT * C4);
    cvt_f16<<<cvt_blocks(Cc * C4), 256, 0, stream>>>(fc2_w + (size_t)l * Cc * C4, wbuf, Cc * C4);
    gemm_nt_res<<<staged_blocks(BT, Cc), 128, 0, stream>>>(
        hid16, C4, wbuf, C4, fc2_b + (size_t)l * Cc, x, x, Cc, BT, Cc, C4);
  }

  // --- final LN + distance head ---
  layernorm_rows<<<128, 256, 0, stream>>>(x, lnf_w, lnf_b, h16, BT);
  rowsumsq<float><<<(Vv + 7) / 8, 256, 0, stream>>>(wte, ww, Vv);
  rowsumsq<h16_t><<<(BT + 7) / 8, 256, 0, stream>>>(h16, xx, BT);
  gemm_dist<<<staged_blocks(BT, Vv), 128, 0, stream>>>(h16, wte16, ww, xx, out);
  row_min<<<BT, 256, 0, stream>>>(out, rmin, Vv);
  dist_pow<<<4096, 256, 0, stream>>>(out, rmin);
  row_sum<<<BT, 256, 0, stream>>>(out, rsum, Vv);
  logits_k<<<4096, 256, 0, stream>>>(out, rsum);
}
